// CustomRNN_49667001811237
// MI455X (gfx1250) — matrix operand order repaired and checked by reading, not compiled, not run on hardware
//
#include <hip/hip_runtime.h>
#include <hip/hip_bf16.h>
#include <stdint.h>

#define B_TOT   256
#define T_STEPS 1000
#define NREC    256
#define NI      3
#define NO      3
#define ALPHA_C 0.2f
#define NSCALE  0.031622776601683794f /* 0.05*sqrt(2*0.2) */

typedef __bf16        bf16x16 __attribute__((ext_vector_type(16)));
typedef float         f32x8   __attribute__((ext_vector_type(8)));
typedef unsigned int  u32x4   __attribute__((ext_vector_type(4)));
typedef int           i32x4   __attribute__((ext_vector_type(4)));
typedef int           i32x8   __attribute__((ext_vector_type(8)));

// Issue one TDM 2D tile load: 16 rows (batch) x 256 floats (N), row stride T*N
// elements in global memory, into LDS at lds_off. Wave-uniform args.
__device__ __forceinline__ void tdm_load_noise_tile(const float* noise, int b0,
                                                    int t, unsigned int lds_off) {
  uint64_t ga = (uint64_t)(uintptr_t)noise +
                (((uint64_t)b0 * T_STEPS + (uint64_t)t) * NREC) * 4ull;
  u32x4 g0;
  g0[0] = 1u;                                              // count=1, user mode
  g0[1] = lds_off;                                         // lds_addr (bytes)
  g0[2] = (unsigned int)ga;                                // global_addr lo
  g0[3] = (unsigned int)((ga >> 32) & 0x1FFFFFFull) | (2u << 30); // hi | type=2
  i32x8 g1;
  g1[0] = (int)(2u << 16);                 // data_size=2 (4 bytes), no mask/pad
  g1[1] = (int)((unsigned)NREC << 16);     // tensor_dim0 = 256 (bits 79:48 lo)
  g1[2] = (int)(16u << 16);                // tensor_dim1 = 16
  g1[3] = (int)((unsigned)NREC << 16);     // tile_dim0 = 256
  g1[4] = 16;                              // tile_dim1 = 16, tile_dim2 = 0
  g1[5] = (int)(T_STEPS * NREC);           // tensor_dim0_stride = 256000 elems
  g1[6] = 0;
  g1[7] = 0;
  i32x4 z4 = {0, 0, 0, 0};                 // 2D tensor: groups 2/3 unused
  i32x8 z8 = {0, 0, 0, 0, 0, 0, 0, 0};
  __builtin_amdgcn_tensor_load_to_lds(g0, g1, z4, z4, z8, 0);
}

__launch_bounds__(512, 1)
__global__ void rnn_scan_kernel(const float* __restrict__ x,
                                const float* __restrict__ noise,
                                const float* __restrict__ W_in,
                                const float* __restrict__ W_rec,
                                const float* __restrict__ W_out_w,
                                const float* __restrict__ W_out_b,
                                const float* __restrict__ bias,
                                float* __restrict__ out) {
  __shared__ __align__(16) float  r_f32[16][NREC];   // state, fp32 (projection)
  __shared__ __align__(16) __bf16 r_b16[16][NREC];   // state, bf16 (WMMA A)
  __shared__ __align__(16) float  nbuf[2][16][NREC]; // double-buffered noise
  __shared__ __align__(16) float  xs[2][16][4];      // staged x_t (3 used)

  const int tid   = threadIdx.x;
  const int wv    = tid >> 5;         // wave 0..15 -> N-tile
  const int ln    = tid & 31;
  const int lhalf = (ln >> 4);        // lane group
  const int l16   = ln & 15;
  const int b0    = blockIdx.x * 16;  // batch tile

  const int ncol = wv * 16 + l16;     // this lane's output column

  // --- time-invariant per-lane preloads -----------------------------------
  const float wi0 = W_in[ncol * NI + 0];
  const float wi1 = W_in[ncol * NI + 1];
  const float wi2 = W_in[ncol * NI + 2];
  const float bcl = bias[ncol];

  float wo[NO][8];                    // output projection weights
#pragma unroll
  for (int o = 0; o < NO; ++o)
#pragma unroll
    for (int j = 0; j < 8; ++j) wo[o][j] = W_out_w[o * NREC + ln + 32 * j];
  const float ob0 = W_out_b[0], ob1 = W_out_b[1], ob2 = W_out_b[2];

  // B fragments (W_rec^T) held in registers across the whole scan.
  // B k order (same as A): lanes 0-15 hold col n, K=kb..kb+15; lanes 16-31 K=kb+16..31.
  const int koff = lhalf ? 16 : 0;
  bf16x16 Bf[8];
#pragma unroll
  for (int kk = 0; kk < 8; ++kk) {
    const float* src = &W_rec[(size_t)ncol * NREC + kk * 32 + koff];
    bf16x16 v;
#pragma unroll
    for (int e = 0; e < 16; ++e) v[e] = (__bf16)src[e];
    Bf[kk] = v;
  }

  // --- init state, stage t=0 inputs ---------------------------------------
  for (int i = tid; i < 16 * NREC; i += 512) {
    r_f32[i >> 8][i & 255] = 0.0f;
    r_b16[i >> 8][i & 255] = (__bf16)0.0f;
  }
  if (tid < 48) {
    int m = tid / 3, i = tid % 3;
    xs[0][m][i] = x[((size_t)(b0 + m) * T_STEPS + 0) * NI + i];
  }
  if (wv == 0)
    tdm_load_noise_tile(noise, b0, 0, (unsigned int)(uintptr_t)&nbuf[0][0][0]);
  __syncthreads();

  const int m_a   = l16;              // WMMA A-operand row for this lane
  const int abase = lhalf ? 16 : 0;   // A K-offset per lane group

  for (int t = 0; t < T_STEPS; ++t) {
    const int cur = t & 1, nxt = cur ^ 1;

    // Prefetch noise tile for t+1 (clamped; keeps TENSORcnt invariant = 2).
    if (wv == 0) {
      int tn = (t + 1 < T_STEPS) ? (t + 1) : (T_STEPS - 1);
      tdm_load_noise_tile(noise, b0, tn,
                          (unsigned int)(uintptr_t)&nbuf[nxt][0][0]);
    }

    // --- recurrent GEMM tile: acc = r_t @ W_rec^T (16x16, K=256) ----------
    // Load all A fragments first (one dscnt wait), then 8 back-to-back WMMAs.
    union { bf16x16 v; uint4 q[2]; } af[8];
#pragma unroll
    for (int kk = 0; kk < 8; ++kk) {
      af[kk].q[0] = *(const uint4*)&r_b16[m_a][kk * 32 + abase];
      af[kk].q[1] = *(const uint4*)&r_b16[m_a][kk * 32 + abase + 8];
    }
    f32x8 acc = {};
#pragma unroll
    for (int kk = 0; kk < 8; ++kk) {
      acc = __builtin_amdgcn_wmma_f32_16x16x32_bf16(
          false, af[kk].v, false, Bf[kk], (short)0, acc, false, false);
    }

    // Oldest TDM (noise for step t) must be complete before the epilogue.
    // Waves 1..15 have TENSORcnt==0, so this is a no-op for them.
    __builtin_amdgcn_s_wait_tensorcnt(1);
    __syncthreads();

    // --- epilogue: drive + leaky ReLU update, write new state -------------
    if (tid < 48) {                   // stage x for t+1
      int m = tid / 3, i = tid % 3;
      int tn = (t + 1 < T_STEPS) ? (t + 1) : t;
      xs[nxt][m][i] = x[((size_t)(b0 + m) * T_STEPS + tn) * NI + i];
    }
#pragma unroll
    for (int v = 0; v < 8; ++v) {
      int m = v + (lhalf ? 8 : 0);    // C/D layout: lanes16-31 hold M=8..15
      float u = xs[cur][m][0] * wi0 + xs[cur][m][1] * wi1 +
                xs[cur][m][2] * wi2 + bcl + NSCALE * nbuf[cur][m][ncol];
      float pre = acc[v] + u;
      float rl  = pre > 0.0f ? pre : 0.0f;
      float rn  = (1.0f - ALPHA_C) * r_f32[m][ncol] + ALPHA_C * rl;
      r_f32[m][ncol] = rn;
      r_b16[m][ncol] = (__bf16)rn;
    }
    __syncthreads();

    // --- output projection: out[b,t,:] = r_t1 @ W_out^T + b ---------------
    {
      const int m = wv;               // one batch row per wave
      float p0 = 0.0f, p1 = 0.0f, p2 = 0.0f;
#pragma unroll
      for (int j = 0; j < 8; ++j) {
        float rv = r_f32[m][ln + 32 * j];
        p0 += rv * wo[0][j];
        p1 += rv * wo[1][j];
        p2 += rv * wo[2][j];
      }
#pragma unroll
      for (int off = 16; off > 0; off >>= 1) {
        p0 += __shfl_down(p0, off, 32);
        p1 += __shfl_down(p1, off, 32);
        p2 += __shfl_down(p2, off, 32);
      }
      if (ln == 0) {
        size_t ob = ((size_t)(b0 + m) * T_STEPS + t) * NO;
        out[ob + 0] = p0 + ob0;
        out[ob + 1] = p1 + ob1;
        out[ob + 2] = p2 + ob2;
      }
    }
    // next iteration's barrier (after its WMMA) orders projection reads
    // against the next epilogue's r_f32 writes.
  }
}

extern "C" void kernel_launch(void* const* d_in, const int* in_sizes, int n_in,
                              void* d_out, int out_size, void* d_ws,
                              size_t ws_size, hipStream_t stream) {
  (void)in_sizes; (void)n_in; (void)out_size; (void)d_ws; (void)ws_size;
  const float* x     = (const float*)d_in[0];
  const float* noise = (const float*)d_in[1];
  const float* W_in  = (const float*)d_in[2];
  const float* W_rec = (const float*)d_in[3];
  const float* Wow   = (const float*)d_in[4];
  const float* Wob   = (const float*)d_in[5];
  const float* bias  = (const float*)d_in[6];
  float* out = (float*)d_out;

  rnn_scan_kernel<<<dim3(B_TOT / 16), dim3(512), 0, stream>>>(
      x, noise, W_in, W_rec, Wow, Wob, bias, out);
}
